// VQ_Many2One_59691455480150
// MI455X (gfx1250) — matrix operand order repaired and checked by reading, not compiled, not run on hardware
//
#include <hip/hip_runtime.h>
#include <hip/hip_bf16.h>

typedef __attribute__((ext_vector_type(16))) _Float16 v16h;
typedef __attribute__((ext_vector_type(8)))  float    v8f;

#define T_SEQ  256
#define BATCH  256
#define HDIM   512
#define DDIM   80
#define KT_H   16            // k-tiles covering h (512 = 16*32)
#define KT_X   3             // k-tiles covering x (80 -> pad 96 = 3*32)
#define KT_TOT 19
#define NPACK  (8*4*4*KT_TOT*512)   // 1,245,184 halves per LSTM weight pack
#define NFC    (5*16*512)           // 40,960 halves for fc pack

// pack strides (halves)
#define STRIDE_G   (4*KT_TOT*512)      // 38,912
#define STRIDE_NT  (KT_TOT*512)        //  9,728
#define STRIDE_W   (4*STRIDE_G)        // 155,648

// d_out layout: recon [B,T,D] f32, then vq_loss f32, then idx[B] (int bits)
#define RECON_ELEMS ((size_t)BATCH * T_SEQ * DDIM)
#define LOSS_OFF    RECON_ELEMS
#define IDX_OFF     (RECON_ELEMS + 1)

// workspace byte offsets
#define WS_ENC_PACK 0
#define WS_DEC_PACK (WS_ENC_PACK + NPACK*2)
#define WS_FC_PACK  (WS_DEC_PACK + NPACK*2)
#define WS_ENC_BIAS (WS_FC_PACK + NFC*2)
#define WS_DEC_BIAS (WS_ENC_BIAS + 2048*4)
#define WS_FCB      (WS_DEC_BIAS + 2048*4)
#define WS_Z        (WS_FCB + 96*4)
#define WS_QUANT    (WS_Z + (size_t)BATCH*HDIM*4)

// fast sigmoid/tanh: raw v_rcp_f32 instead of IEEE divide expansion
__device__ __forceinline__ float sigf(float x) {
    return __builtin_amdgcn_rcpf(1.0f + __expf(-x));
}
__device__ __forceinline__ float tanh_fast(float x) {
    return 1.0f - 2.0f * __builtin_amdgcn_rcpf(__expf(2.0f * x) + 1.0f);
}

// ---------------------------------------------------------------------------
// Prep: repack weights into CDNA5 WMMA B-fragment layout (f16), fold biases.
// B-frag layout (16x16x32, B 32x16): lane L holds N = L%16; halves h=0..15
// hold K = (L<16 ? 0 : 8) + (h<8 ? h : h+8) within the 32-deep k-tile; 32B contiguous/lane.
// ---------------------------------------------------------------------------
__global__ __launch_bounds__(256) void pack_kernel(
    const float* __restrict__ eWih, const float* __restrict__ eWhh,
    const float* __restrict__ eBih, const float* __restrict__ eBhh,
    const float* __restrict__ dWih, const float* __restrict__ dWhh,
    const float* __restrict__ dBih, const float* __restrict__ dBhh,
    const float* __restrict__ fcW,  const float* __restrict__ fcb,
    _Float16* __restrict__ encP, _Float16* __restrict__ decP,
    _Float16* __restrict__ fcP,
    float* __restrict__ encB, float* __restrict__ decB, float* __restrict__ fcBp,
    float* __restrict__ outLoss)
{
    const int p = blockIdx.x * 256 + threadIdx.x;
    if (p < NPACK) {
        const int pos  = p & 511,  frag = p >> 9;
        const int lane = pos >> 4, hf   = pos & 15;
        const int kt = frag % KT_TOT;
        int rem = frag / KT_TOT;
        const int nt = rem & 3;  rem >>= 2;
        const int g  = rem & 3;
        const int w  = rem >> 2;                 // wave 0..7
        const int n  = lane & 15;
        const int j  = g*HDIM + w*64 + nt*16 + n;   // gate row (i,f,g,o order)
        const int k  = kt*32 + ((lane < 16) ? 0 : 8) + ((hf < 8) ? hf : hf + 8);
        float ev, dv;
        if (kt < KT_H) {
            ev = eWhh[(size_t)j*HDIM + k];
            dv = dWhh[(size_t)j*HDIM + k];
        } else {
            const int d = k - HDIM;              // 0..95
            ev = (d < DDIM) ? eWih[(size_t)j*DDIM + d] : 0.0f;
            dv = (d < DDIM) ? dWih[(size_t)j*DDIM + d] : 0.0f;
        }
        encP[p] = (_Float16)ev;
        decP[p] = (_Float16)dv;
    }
    if (p < NFC) {
        const int pos  = p & 511,  frag = p >> 9;
        const int lane = pos >> 4, hf   = pos & 15;
        const int kt  = frag & 15, nt = frag >> 4;      // nt 0..4
        const int row = nt*16 + (lane & 15);            // output dim 0..79
        const int k   = kt*32 + ((lane < 16) ? 0 : 8) + ((hf < 8) ? hf : hf + 8);
        fcP[p] = (_Float16)fcW[(size_t)row*HDIM + k];
    }
    if (p < 2048) { encB[p] = eBih[p] + eBhh[p]; decB[p] = dBih[p] + dBhh[p]; }
    if (p < 96)   { fcBp[p] = (p < DDIM) ? fcb[p] : 0.0f; }
    if (p == 0)   { *outLoss = 0.0f; }
}

// map A-frag (lane,half) -> k_local
__device__ __forceinline__ int afrag_klocal(int ln, int hf) {
    return ((ln < 16) ? 0 : 8) + ((hf < 8) ? hf : hf + 8);
}

// ---------------------------------------------------------------------------
// One LSTM step for a 16-row batch tile. Wave w owns h-columns [64w,64w+64):
// 4 n-tiles x 4 gates x 19 k-tiles of v_wmma_f32_16x16x32_f16. c in regs.
// kt loop unrolled x2: ~160 live VGPRs, deeper load pipelining, no spills.
// ---------------------------------------------------------------------------
__device__ __forceinline__ void lstm_gates_tile(
    const _Float16* __restrict__ afragCur,   // [KT_TOT][512]
    _Float16*       __restrict__ afragNxt,
    const _Float16* __restrict__ packW,      // pack base (whole WG)
    const float*    __restrict__ bias,
    int wv, int lane, v8f c[4])
{
    const int ncol    = lane & 15;
    const int colbase = wv * 64;
    const int mhi     = (lane >> 4) * 8;
    const _Float16* pA = afragCur + lane * 16;                    // + kt*512
    const _Float16* pW = packW + (size_t)wv * STRIDE_W + lane*16; // + g/nt/kt strides
    #pragma unroll
    for (int nt = 0; nt < 4; ++nt) {
        v8f acc[4];
        #pragma unroll
        for (int g = 0; g < 4; ++g) {
            const float bv = bias[g*HDIM + colbase + nt*16 + ncol];
            #pragma unroll
            for (int e = 0; e < 8; ++e) acc[g][e] = bv;
        }
        const _Float16* pWnt = pW + nt * STRIDE_NT;
        #pragma unroll 2
        for (int kt = 0; kt < KT_TOT; ++kt) {
            const v16h a = *(const v16h*)(pA + kt*512);
            #pragma unroll
            for (int g = 0; g < 4; ++g) {
                const v16h b = *(const v16h*)(pWnt + g*STRIDE_G + kt*512);
                acc[g] = __builtin_amdgcn_wmma_f32_16x16x32_f16(
                    false, a, false, b, (short)0, acc[g], false, false);
            }
        }
        // elementwise LSTM cell update (c persistent in registers)
        const int kh    = colbase + nt*16 + ncol;
        const int kt2   = kh >> 5, kl = kh & 31;
        const int lofs  = (kl & 8) ? 16 : 0;
        const int half2 = (kl & 7) + ((kl & 16) ? 8 : 0);
        #pragma unroll
        for (int e = 0; e < 8; ++e) {
            const float nc = sigf(acc[1][e]) * c[nt][e]
                           + sigf(acc[0][e]) * tanh_fast(acc[2][e]);
            c[nt][e] = nc;
            const float h = sigf(acc[3][e]) * tanh_fast(nc);
            // scatter h (f16) into next buffer's A-fragment layout:
            //  kl 0..7 -> lane m, half kl ; 8..15 -> lane m+16, half kl-8
            //  16..23 -> lane m, half kl-8; 24..31 -> lane m+16, half kl-16
            afragNxt[kt2*512 + (e + mhi + lofs)*16 + half2] = (_Float16)h;
        }
    }
}

// ---------------------------------------------------------------------------
// Encoder: 16 blocks x 256 threads; each block runs 256 steps for 16 rows.
// ---------------------------------------------------------------------------
__global__ __launch_bounds__(256) void enc_kernel(
    const float* __restrict__ traj, const _Float16* __restrict__ packW,
    const float* __restrict__ bias, float* __restrict__ z)
{
    __shared__ __align__(32) _Float16 afrag[2][KT_TOT][512];
    const int tid  = threadIdx.x;
    const int lane = tid & 31;
    const int wv   = tid >> 5;
    const int b0   = blockIdx.x * 16;

    v8f c[4];
    #pragma unroll
    for (int nt = 0; nt < 4; ++nt)
        #pragma unroll
        for (int e = 0; e < 8; ++e) c[nt][e] = 0.0f;

    for (int p = tid; p < KT_H*512; p += 256)          // h0 = 0
        afrag[0][p >> 9][p & 511] = (_Float16)0.0f;

    int cur = 0;
    #pragma unroll 1
    for (int t = 0; t < T_SEQ; ++t) {
        // stage x_t into A-frag k-tiles 16..18 of current buffer
        for (int p = tid; p < KT_X*512; p += 256) {
            const int fr = p >> 9, pos = p & 511;
            const int ln = pos >> 4, hf = pos & 15;
            const int m  = ln & 15;
            const int d  = fr*32 + afrag_klocal(ln, hf);
            float v = 0.0f;
            if (d < DDIM) v = traj[((size_t)(b0 + m)*T_SEQ + t)*DDIM + d];
            afrag[cur][KT_H + fr][pos] = (_Float16)v;
        }
        __syncthreads();
        lstm_gates_tile(&afrag[cur][0][0], &afrag[cur ^ 1][0][0],
                        packW, bias, wv, lane, c);
        cur ^= 1;
    }
    // emit z = final h (read back from LDS fragments)
    __syncthreads();
    for (int p = tid; p < KT_H*512; p += 256) {
        const int kt = p >> 9, pos = p & 511;
        const int ln = pos >> 4, hf = pos & 15;
        const int m  = ln & 15;
        const int k  = kt*32 + afrag_klocal(ln, hf);
        z[(size_t)(b0 + m)*HDIM + k] = (float)afrag[cur][kt][pos];
    }
}

// ---------------------------------------------------------------------------
// Vector quantizer: one block per batch row; argmin + quantized + loss.
// ---------------------------------------------------------------------------
__global__ __launch_bounds__(256) void vq_kernel(
    const float* __restrict__ z, const float* __restrict__ emb,
    float* __restrict__ quant, float* __restrict__ outBase)
{
    __shared__ float zrow[HDIM];
    __shared__ float best[256];
    __shared__ int   bidx[256];
    const int b = blockIdx.x, tid = threadIdx.x;
    for (int i = tid; i < HDIM; i += 256) zrow[i] = z[(size_t)b*HDIM + i];
    __syncthreads();

    float bd = 3.4e38f; int bi = 0;
    for (int e = tid; e < 512; e += 256) {
        const float* er = emb + (size_t)e*HDIM;
        float dot = 0.0f, nn = 0.0f;
        for (int k = 0; k < HDIM; ++k) { const float w = er[k]; dot += w*zrow[k]; nn += w*w; }
        const float d = nn - 2.0f*dot;            // + |z|^2 (const per row)
        if (d < bd) { bd = d; bi = e; }
    }
    best[tid] = bd; bidx[tid] = bi;
    __syncthreads();
    for (int s = 128; s > 0; s >>= 1) {
        if (tid < s) {
            if (best[tid+s] < best[tid] ||
               (best[tid+s] == best[tid] && bidx[tid+s] < bidx[tid])) {
                best[tid] = best[tid+s]; bidx[tid] = bidx[tid+s];
            }
        }
        __syncthreads();
    }
    const int idx = bidx[0];
    if (tid == 0) ((int*)outBase)[IDX_OFF + b] = idx;

    float lsum = 0.0f;
    for (int i = tid; i < HDIM; i += 256) {
        const float q = emb[(size_t)idx*HDIM + i];
        quant[(size_t)b*HDIM + i] = q;
        const float dd = q - zrow[i];
        lsum += dd*dd;
    }
    __syncthreads();
    best[tid] = lsum;
    __syncthreads();
    for (int s = 128; s > 0; s >>= 1) {
        if (tid < s) best[tid] += best[tid+s];
        __syncthreads();
    }
    if (tid == 0)
        atomicAdd(&outBase[LOSS_OFF], best[0] * (1.25f / ((float)BATCH * (float)HDIM)));
}

// ---------------------------------------------------------------------------
// Decoder: like encoder but h0 = quantized, x0 = 0; per step also computes
// y_t = h_t @ fc_W^T + fc_b (WMMA, waves 0..4), stores recon, feeds x_{t+1}.
// ---------------------------------------------------------------------------
__global__ __launch_bounds__(256) void dec_kernel(
    const float* __restrict__ quant, const _Float16* __restrict__ packW,
    const float* __restrict__ bias,  const _Float16* __restrict__ fcP,
    const float* __restrict__ fcb,   float* __restrict__ recon)
{
    __shared__ __align__(32) _Float16 afrag[2][KT_TOT][512];
    const int tid  = threadIdx.x;
    const int lane = tid & 31;
    const int wv   = tid >> 5;
    const int b0   = blockIdx.x * 16;

    for (int p = tid; p < KT_H*512; p += 256) {        // h0 = quantized
        const int kt = p >> 9, pos = p & 511;
        const int ln = pos >> 4, hf = pos & 15;
        const int m  = ln & 15;
        const int k  = kt*32 + afrag_klocal(ln, hf);
        afrag[0][kt][pos] = (_Float16)quant[(size_t)(b0 + m)*HDIM + k];
    }
    for (int p = tid; p < 2*KT_X*512; p += 256) {      // x regions (both bufs) = 0
        const int buf = p / (KT_X*512);
        const int r   = p % (KT_X*512);
        afrag[buf][KT_H + (r >> 9)][r & 511] = (_Float16)0.0f;
    }

    v8f c[4];
    #pragma unroll
    for (int nt = 0; nt < 4; ++nt)
        #pragma unroll
        for (int e = 0; e < 8; ++e) c[nt][e] = 0.0f;

    int cur = 0;
    #pragma unroll 1
    for (int t = 0; t < T_SEQ; ++t) {
        __syncthreads();
        lstm_gates_tile(&afrag[cur][0][0], &afrag[cur ^ 1][0][0],
                        packW, bias, wv, lane, c);
        __syncthreads();                               // h_t frags visible
        if (wv < 5) {                                  // y = h @ fcW^T + fcb
            const int ncol = lane & 15;
            const int col  = wv*16 + ncol;             // 0..79
            v8f y;
            const float bv = fcb[col];
            #pragma unroll
            for (int e = 0; e < 8; ++e) y[e] = bv;
            const _Float16* pA  = &afrag[cur ^ 1][0][0] + lane*16;
            const _Float16* pFc = fcP + (size_t)wv*16*512 + lane*16;
            #pragma unroll 2
            for (int kt = 0; kt < KT_H; ++kt) {
                const v16h a = *(const v16h*)(pA + kt*512);
                const v16h b = *(const v16h*)(pFc + kt*512);
                y = __builtin_amdgcn_wmma_f32_16x16x32_f16(
                    false, a, false, b, (short)0, y, false, false);
            }
            const int kt2   = KT_H + (col >> 5);
            const int kl    = col & 31;
            const int lofs  = (kl & 8) ? 16 : 0;
            const int half2 = (kl & 7) + ((kl & 16) ? 8 : 0);
            const int mhi   = (lane >> 4) * 8;
            #pragma unroll
            for (int e = 0; e < 8; ++e) {
                const int m = e + mhi;
                recon[((size_t)(b0 + m)*T_SEQ + t)*DDIM + col] = y[e];
                afrag[cur ^ 1][kt2][(m + lofs)*16 + half2] = (_Float16)y[e]; // x_{t+1}
            }
        }
        cur ^= 1;
    }
}

// ---------------------------------------------------------------------------
extern "C" void kernel_launch(void* const* d_in, const int* in_sizes, int n_in,
                              void* d_out, int out_size, void* d_ws, size_t ws_size,
                              hipStream_t stream)
{
    const float* traj = (const float*)d_in[0];
    // d_in[1] = seq_len (256, hardcoded)
    const float* eWih = (const float*)d_in[2];
    const float* eWhh = (const float*)d_in[3];
    const float* eBih = (const float*)d_in[4];
    const float* eBhh = (const float*)d_in[5];
    const float* emb  = (const float*)d_in[6];
    const float* dWih = (const float*)d_in[7];
    const float* dWhh = (const float*)d_in[8];
    const float* dBih = (const float*)d_in[9];
    const float* dBhh = (const float*)d_in[10];
    const float* fcW  = (const float*)d_in[11];
    const float* fcb  = (const float*)d_in[12];

    char* ws = (char*)d_ws;
    _Float16* encP  = (_Float16*)(ws + WS_ENC_PACK);
    _Float16* decP  = (_Float16*)(ws + WS_DEC_PACK);
    _Float16* fcP   = (_Float16*)(ws + WS_FC_PACK);
    float*    encB  = (float*)(ws + WS_ENC_BIAS);
    float*    decB  = (float*)(ws + WS_DEC_BIAS);
    float*    fcBp  = (float*)(ws + WS_FCB);
    float*    zbuf  = (float*)(ws + WS_Z);
    float*    qbuf  = (float*)(ws + WS_QUANT);
    float*    out   = (float*)d_out;

    pack_kernel<<<(NPACK + 255)/256, 256, 0, stream>>>(
        eWih, eWhh, eBih, eBhh, dWih, dWhh, dBih, dBhh, fcW, fcb,
        encP, decP, fcP, encB, decB, fcBp, out + LOSS_OFF);

    enc_kernel<<<BATCH/16, 256, 0, stream>>>(traj, encP, encB, zbuf);
    vq_kernel<<<BATCH, 256, 0, stream>>>(zbuf, emb, qbuf, out);
    dec_kernel<<<BATCH/16, 256, 0, stream>>>(qbuf, decP, decB, fcP, fcBp, out);
}
